// MeshConv_transpose_62388694942535
// MI455X (gfx1250) — matrix operand order repaired and checked by reading, not compiled, not run on hardware
//
#include <hip/hip_runtime.h>
#include <hip/hip_bf16.h>

#define B_       8
#define CIN_     64
#define COUT_    64
#define NV_PREV_ 10242
#define NV_      40962
#define NF_      81920
#define KTOT_    256            // CIN*4
#define NTILES_M ((NV_ + 15) / 16)

typedef __attribute__((ext_vector_type(16))) __bf16 v16bf;
typedef __attribute__((ext_vector_type(8)))  float  v8f;

__device__ __forceinline__ unsigned short f2bf(float f) {
    unsigned u = __float_as_uint(f);
    u += 0x7FFFu + ((u >> 16) & 1u);          // round-to-nearest-even
    return (unsigned short)(u >> 16);
}
__device__ __forceinline__ float bf2f(unsigned short s) {
    return __uint_as_float(((unsigned)s) << 16);
}
// xin(b,c,v): x for v < NV_PREV, else 1.0 (padding of upsampled vertices)
__device__ __forceinline__ float xin_val(const float* __restrict__ x, int bc, int v) {
    return (v < NV_PREV_) ? x[(size_t)bc * NV_PREV_ + v] : 1.0f;
}

// ---------------------------------------------------------------------------
// Kernel 0: pre-swizzle coeffs [COUT][CIN][4] -> bf16 WMMA B fragments.
// B operand layout (32x16 bf16, wave32): lane l = (n = l%16 col, h = l/16),
// VGPR r holds K = ks*32 + (r<4 ? 2r + 8h : 16 + 2(r-4) + 8h) (+1 in high half). Stored so the GEMM
// kernel reads 8 consecutive uints per lane: idx = ((nt*8+ks)*32 + l)*8 + r.
// ---------------------------------------------------------------------------
__global__ void prep_coeffs(const float* __restrict__ coeffs, unsigned* __restrict__ Bfrag) {
    int gid = blockIdx.x * blockDim.x + threadIdx.x;
    if (gid >= 4 * 8 * 32 * 8) return;
    int r  = gid & 7;
    int l  = (gid >> 3) & 31;
    int ks = (gid >> 8) & 7;
    int nt = gid >> 11;
    int n = l & 15, h = l >> 4;
    int o   = nt * 16 + n;                       // output channel (N)
    int kk0 = ks * 32 + ((r < 4) ? (2 * r + 8 * h) : (16 + 2 * (r - 4) + 8 * h));          // K index (kk = c*4 + k)
    unsigned short lo = f2bf(coeffs[(size_t)o * KTOT_ + kk0]);
    unsigned short hi = f2bf(coeffs[(size_t)o * KTOT_ + kk0 + 1]);
    Bfrag[gid] = (unsigned)lo | ((unsigned)hi << 16);
}

// ---------------------------------------------------------------------------
// Kernel 1: per-face gradient projected onto EW/NS tangents, packed bf16x2.
// face_out[bc*NF + f] = pack(bf16(ew), bf16(ns))
// ---------------------------------------------------------------------------
__global__ void face_kernel(const float* __restrict__ x,
                            const int*   __restrict__ Gc,
                            const float* __restrict__ Gv,
                            const float* __restrict__ EW,
                            const float* __restrict__ NS,
                            unsigned* __restrict__ face_out) {
    int f  = blockIdx.x * blockDim.x + threadIdx.x;   // face id
    int bc = blockIdx.y;                              // b*CIN + c
    if (f >= NF_) return;
    float ew = 0.f, ns = 0.f;
    #pragma unroll
    for (int d = 0; d < 3; ++d) {
        long base = (long)(3 * f + d) * 3;
        float g = 0.f;
        #pragma unroll
        for (int j = 0; j < 3; ++j)
            g += Gv[base + j] * xin_val(x, bc, Gc[base + j]);
        ew += g * EW[f * 3 + d];
        ns += g * NS[f * 3 + d];
    }
    face_out[(size_t)bc * NF_ + f] = (unsigned)f2bf(ew) | ((unsigned)f2bf(ns) << 16);
}

// ---------------------------------------------------------------------------
// Kernel 2: fused feature build (LDS, bf16) + WMMA GEMM + bias.
// Block = 128 threads = 4 waves; one 16-vertex M-tile per block; wave w owns
// N-tile w (16 of the 64 output channels). K = 256 -> 8 wmma per wave.
// ---------------------------------------------------------------------------
__global__ __launch_bounds__(128) void meshconv_gemm(
        const float*    __restrict__ x,
        const int*      __restrict__ Lc,  const float* __restrict__ Lv,
        const int*      __restrict__ Fc,  const float* __restrict__ Fv,
        const unsigned* __restrict__ face_ewns,
        const unsigned* __restrict__ Bfrag,
        const float*    __restrict__ bias,
        float*          __restrict__ out) {
    __shared__ unsigned short featA[16][KTOT_ + 4];  // bf16 A-tile, padded rows
    __shared__ int   sLc[16 * 7];
    __shared__ float sLv[16 * 7];
    __shared__ int   sFc[16 * 6];
    __shared__ float sFv[16 * 6];

    const int tile = blockIdx.x;       // M tile over vertices
    const int b    = blockIdx.y;       // batch
    const int v0   = tile * 16;
    const int t    = threadIdx.x;

    // --- stage sparse indices for this vertex tile (clamped; invalid rows zeroed later)
    if (t < 112) {
        int v = v0 + t / 7;  if (v >= NV_) v = NV_ - 1;
        int idx = v * 7 + (t % 7);
        sLc[t] = Lc[idx];  sLv[t] = Lv[idx];
    } else if (t < 112 + 96) {
        int q = t - 112;
        int v = v0 + q / 6;  if (v >= NV_) v = NV_ - 1;
        int idx = v * 6 + (q % 6);
        sFc[q] = Fc[idx];  sFv[q] = Fv[idx];
    }
    __syncthreads();

    // --- phase 1: build feat tile [16 rows][256 K] in bf16 (k fast: kk = c*4+k)
    const int vl = t & 15;             // vertex within tile
    const int cb = t >> 4;             // channel base 0..7
    const int vg = v0 + vl;
    const bool valid = (vg < NV_);
    for (int cc = 0; cc < 8; ++cc) {
        int c  = cb + cc * 8;          // 0..63
        int bc = b * CIN_ + c;
        float xv = 0.f, lap = 0.f, gew = 0.f, gns = 0.f;
        if (valid) {
            xv = xin_val(x, bc, vg);
            #pragma unroll
            for (int j = 0; j < 7; ++j)
                lap += sLv[vl * 7 + j] * xin_val(x, bc, sLc[vl * 7 + j]);
            const unsigned* fr = face_ewns + (size_t)bc * NF_;
            #pragma unroll
            for (int j = 0; j < 6; ++j) {
                unsigned p = fr[sFc[vl * 6 + j]];
                float w = sFv[vl * 6 + j];
                gew += w * bf2f((unsigned short)(p & 0xffffu));
                gns += w * bf2f((unsigned short)(p >> 16));
            }
        }
        unsigned* dst = (unsigned*)&featA[vl][c * 4];
        dst[0] = (unsigned)f2bf(xv)  | ((unsigned)f2bf(lap) << 16);
        dst[1] = (unsigned)f2bf(gew) | ((unsigned)f2bf(gns) << 16);
    }
    __syncthreads();

    // --- phase 2: WMMA. wave w -> N-tile w. lane l = (n = l%16, h = l/16).
    const int nt = t >> 5;
    const int l  = t & 31;
    const int n  = l & 15, h = l >> 4;

    v8f acc = {};
    const unsigned* bptr = Bfrag + (size_t)nt * 2048 + (size_t)l * 8;
    #pragma unroll
    for (int ks = 0; ks < 8; ++ks) {
        union { unsigned u[8]; v16bf v; } a, bb;
        #pragma unroll
        for (int r = 0; r < 8; ++r) {
            // 16-bit A 16x32 layout: r<4 -> K = 2r + 8h ; r>=4 -> K = 16 + 2(r-4) + 8h
            int kb = ks * 32 + ((r < 4) ? (2 * r + 8 * h) : (16 + 2 * (r - 4) + 8 * h));
            a.u[r]  = *(const unsigned*)&featA[n][kb];     // row M = l%16
            bb.u[r] = bptr[(size_t)ks * 256 + r];          // contiguous 32B per lane
        }
        acc = __builtin_amdgcn_wmma_f32_16x16x32_bf16(
                  /*neg_a=*/false, a.v, /*neg_b=*/false, bb.v,
                  /*c_mod=*/(short)0, acc, /*reuse_a=*/false, /*reuse_b=*/false);
    }

    // --- bias + store: D layout -> lane(n,h), VGPR r holds (M = r + 8h, N = n)
    const int o = nt * 16 + n;
    const float bv = bias[o];
    const int vbase = v0 + 8 * h;
    float* op = out + ((size_t)b * COUT_ + o) * NV_ + vbase;
    #pragma unroll
    for (int r = 0; r < 8; ++r) {
        if (vbase + r < NV_) op[r] = acc[r] + bv;
    }
}

// ---------------------------------------------------------------------------
// Host launcher
// Input order: x, G_rows, G_cols, G_vals, L_rows, L_cols, L_vals,
//              F2V_rows, F2V_cols, F2V_vals, EW, NS, coeffs, bias
// Workspace: [ face_ewns: B*CIN*NF uints = 167.8 MB ][ Bfrag: 8192 uints ]
// ---------------------------------------------------------------------------
extern "C" void kernel_launch(void* const* d_in, const int* in_sizes, int n_in,
                              void* d_out, int out_size, void* d_ws, size_t ws_size,
                              hipStream_t stream) {
    const float* x      = (const float*)d_in[0];
    const int*   Gc     = (const int*)  d_in[2];
    const float* Gv     = (const float*)d_in[3];
    const int*   Lc     = (const int*)  d_in[5];
    const float* Lv     = (const float*)d_in[6];
    const int*   Fc     = (const int*)  d_in[8];
    const float* Fv     = (const float*)d_in[9];
    const float* EW     = (const float*)d_in[10];
    const float* NS     = (const float*)d_in[11];
    const float* coeffs = (const float*)d_in[12];
    const float* bias   = (const float*)d_in[13];
    float*       out    = (float*)d_out;

    unsigned* face  = (unsigned*)d_ws;
    unsigned* Bfrag = face + (size_t)B_ * CIN_ * NF_;

    prep_coeffs<<<dim3(32), dim3(256), 0, stream>>>(coeffs, Bfrag);
    face_kernel<<<dim3(NF_ / 256, B_ * CIN_), dim3(256), 0, stream>>>(x, Gc, Gv, EW, NS, face);
    meshconv_gemm<<<dim3(NTILES_M, B_), dim3(128), 0, stream>>>(
        x, Lc, Lv, Fc, Fv, face, Bfrag, bias, out);
}
